// MultiLayerTimeAwareGNN_2748779069857
// MI455X (gfx1250) — matrix operand order repaired and checked by reading, not compiled, not run on hardware
//
#include <hip/hip_runtime.h>
#include <hip/hip_bf16.h>
#include <math.h>

// ---------------------------------------------------------------------------
// MultiLayerTimeAwareGNN (TGN TransformerConv x3) for gfx1250 / MI455X.
// GEMMs: v_wmma_f32_16x16x32_bf16, 32x64 tile per wave (8 WMMA / K-chunk),
// all fragment/store addresses folded to base + immediate offsets (NC=256
// compile-time). A fragments = 2x b128 contiguous loads; B pre-swizzled to
// fragment order so B fragments = 2x b128. Attention softmax: encoded-u32
// atomic max + exp/atomic-add scatter; node-side buffers (~105MB) live in
// the 192MB L2 so src/dst gathers and scatter atomics stay on-chip.
// ---------------------------------------------------------------------------

typedef __bf16 bf16_t;
typedef __attribute__((ext_vector_type(16))) __bf16 v16bf;
typedef __attribute__((ext_vector_type(8)))  __bf16 v8bf;
typedef __attribute__((ext_vector_type(8)))  float  v8f;

#define N_NODES 20000
#define N_EDGES 320000
#define MEM_D   128
#define IN_D    256
#define N_HEAD  2
#define C_D     128
#define HC_D    256      // H*C
#define MSG_D   100
#define TD_D    100
#define ED_D    200      // edge feature dim
#define EDP_D   224      // padded to multiple of 32 for WMMA K
#define N_LAYER 3

__device__ __forceinline__ v16bf cat8(v8bf lo, v8bf hi) {
  return __builtin_shufflevector(lo, hi, 0, 1, 2, 3, 4, 5, 6, 7,
                                 8, 9, 10, 11, 12, 13, 14, 15);
}

// ---------------------------------------------------------------------------
// Elementwise f32 -> bf16 (row-major, used for A-side activations)
// ---------------------------------------------------------------------------
__global__ void f32_to_bf16_kernel(const float* __restrict__ src,
                                   bf16_t* __restrict__ dst, int n) {
  int i = blockIdx.x * blockDim.x + threadIdx.x;
  if (i < n) dst[i] = (bf16_t)src[i];
}

// ---------------------------------------------------------------------------
// Pack weight W[Ksrc x Nc] (f32 row-major) into WMMA-B fragment order:
// out[(kc*Nc + n)*32 + t] = W[kc*32+t][n]  (zero beyond Ksrc rows).
// Each lane's B fragment is two 16-byte runs (t = 8h+[0,8), 16+8h+[0,8)) -> 2x global_load_b128.
// ---------------------------------------------------------------------------
__global__ void pack_weight_kernel(const float* __restrict__ W,
                                   bf16_t* __restrict__ out,
                                   int Ksrc, int Kpad, int Nc) {
  int idx = blockIdx.x * blockDim.x + threadIdx.x;
  if (idx >= Kpad * Nc) return;
  int t  = idx & 31;
  int n  = (idx >> 5) % Nc;
  int kc = idx / (32 * Nc);
  int k  = kc * 32 + t;
  float v = (k < Ksrc) ? W[(size_t)k * Nc + n] : 0.f;
  out[idx] = (bf16_t)v;
}

// ---------------------------------------------------------------------------
// Edge features: eattr[e] = [cos(rel_t * w + b) (100) | msg (100) | 0-pad]
// ---------------------------------------------------------------------------
__global__ void edge_feat_kernel(const int* __restrict__ ei,
                                 const float* __restrict__ last_update,
                                 const float* __restrict__ t,
                                 const float* __restrict__ msg,
                                 const float* __restrict__ tw,
                                 const float* __restrict__ tb,
                                 bf16_t* __restrict__ eattr) {
  int e = blockIdx.x;
  int c = threadIdx.x;               // 0..223
  int s = ei[e];
  float rel = last_update[s] - t[e];
  float v;
  if (c < TD_D)       v = cosf(rel * tw[c] + tb[c]);
  else if (c < ED_D)  v = msg[(size_t)e * MSG_D + (c - TD_D)];
  else                v = 0.f;
  eattr[(size_t)e * EDP_D + c] = (bf16_t)v;
}

// ---------------------------------------------------------------------------
// WMMA GEMM: D[M x 256] = A[M x K](bf16 row-major, lda) @ Bp(packed) + bias.
// One wave computes a 32(M) x 64(N) tile: 2 A frags x 4 B frags -> 8 WMMAs
// per 32-wide K-chunk; K fully unrolled. All loads/stores are single-base +
// compile-time immediate offsets (NC fixed at 256).
// ---------------------------------------------------------------------------
template <int KDIM>
__global__ void gemm_wmma_bf16(const bf16_t* __restrict__ A, int lda,
                               const bf16_t* __restrict__ Bp,
                               const float* __restrict__ bias,
                               float* __restrict__ Cf,
                               bf16_t* __restrict__ Cb,
                               int M) {
  constexpr int NC = 256;
  constexpr int TILES_N = NC / 64;           // 4 column groups of 64
  int wid  = (int)((blockIdx.x * blockDim.x + threadIdx.x) >> 5);
  int lane = threadIdx.x & 31;
  int total = (M >> 5) * TILES_N;
  if (wid >= total) return;                  // uniform per wave
  int tm = wid / TILES_N;
  int tn = wid % TILES_N;
  int nrow = lane & 15;                      // A row in tile / B,C column
  int hi   = lane >> 4;

  v8f acc[2][4] = {};
  // A fragment: per-lane contiguous runs  k = kk + hi*8 + [0,8) and +16.
  const char* a0p = (const char*)(A + (size_t)(tm * 32 + nrow) * lda + (hi << 3));
  const char* a1p = a0p + (size_t)lda * 32;  // +16 rows (2 bytes/elem)
  // B fragment base: elements (n0*32 + hi*8); per (kc,j): +kc*16384 + j*1024 B
  int n0 = tn * 64 + nrow;
  const char* bbase = (const char*)Bp + ((size_t)n0 << 6) + (hi << 4);

#pragma unroll
  for (int kk = 0; kk < KDIM; kk += 32) {
    const int kc = kk >> 5;
    v16bf a0 = cat8(*(const v8bf*)(a0p + kk * 2),
                    *(const v8bf*)(a0p + kk * 2 + 32));
    v16bf a1 = cat8(*(const v8bf*)(a1p + kk * 2),
                    *(const v8bf*)(a1p + kk * 2 + 32));
#pragma unroll
    for (int j = 0; j < 4; ++j) {
      const char* bp = bbase + kc * (NC * 64) + j * 1024;
      v16bf b = cat8(*(const v8bf*)bp, *(const v8bf*)(bp + 32));
      acc[0][j] = __builtin_amdgcn_wmma_f32_16x16x32_bf16(
          false, a0, false, b, (short)0, acc[0][j], false, false);
      acc[1][j] = __builtin_amdgcn_wmma_f32_16x16x32_bf16(
          false, a1, false, b, (short)0, acc[1][j], false, false);
    }
  }

  float bv[4] = {0.f, 0.f, 0.f, 0.f};
  if (bias) {
#pragma unroll
    for (int j = 0; j < 4; ++j) bv[j] = bias[tn * 64 + j * 16 + nrow];
  }
  // C/D layout: VGPR r -> row = tm*32 + mh*16 + hi*8 + r, col = tn*64+j*16+nrow
  size_t cbase = (size_t)(tm * 32 + hi * 8) * NC + (size_t)(tn * 64 + nrow);
  if (Cf) {
    char* cp = (char*)(Cf + cbase);
#pragma unroll
    for (int mh = 0; mh < 2; ++mh)
#pragma unroll
      for (int j = 0; j < 4; ++j)
#pragma unroll
        for (int r = 0; r < 8; ++r)
          *(float*)(cp + (mh * 16 + r) * (NC * 4) + j * 64) = acc[mh][j][r] + bv[j];
  }
  if (Cb) {
    char* cp = (char*)(Cb + cbase);
#pragma unroll
    for (int mh = 0; mh < 2; ++mh)
#pragma unroll
      for (int j = 0; j < 4; ++j)
#pragma unroll
        for (int r = 0; r < 8; ++r)
          *(bf16_t*)(cp + (mh * 16 + r) * (NC * 2) + j * 32) =
              (bf16_t)(acc[mh][j][r] + bv[j]);
  }
}

// ---------------------------------------------------------------------------
// Per-layer segment-state init: S=0, denom=0, amax=enc(-inf)
// ---------------------------------------------------------------------------
__global__ void init_seg_kernel(float* __restrict__ S,
                                float* __restrict__ denom,
                                unsigned* __restrict__ amax) {
  int i = blockIdx.x * blockDim.x + threadIdx.x;
  if (i < N_NODES * HC_D) S[i] = 0.f;
  if (i < N_NODES * N_HEAD) { denom[i] = 0.f; amax[i] = 0x007FFFFFu; }
}

__device__ __forceinline__ unsigned enc_f32(float x) {
  unsigned u = __float_as_uint(x);
  return (u & 0x80000000u) ? ~u : (u | 0x80000000u);
}
__device__ __forceinline__ float dec_f32(unsigned u) {
  return __uint_as_float((u & 0x80000000u) ? (u & 0x7FFFFFFFu) : ~u);
}

// ---------------------------------------------------------------------------
// Pass 1: alpha[e,h] = q[dst,h,:]·(k[src,h,:]+e[e,h,:]) / sqrt(C); atomic max.
// One wave per edge; ch = j*32 + lane (coalesced); j<4 -> head0, j>=4 -> head1.
// ---------------------------------------------------------------------------
__global__ void attn_alpha_kernel(const int* __restrict__ ei,
                                  const float* __restrict__ q,
                                  const float* __restrict__ kmat,
                                  const bf16_t* __restrict__ eproj,
                                  float* __restrict__ alpha,
                                  unsigned* __restrict__ amax) {
  int wid  = (int)((blockIdx.x * blockDim.x + threadIdx.x) >> 5);
  int lane = threadIdx.x & 31;
  if (wid >= N_EDGES) return;
  int s = ei[wid], d = ei[N_EDGES + wid];
  const float*  qr = q     + (size_t)d * HC_D;
  const float*  kr = kmat  + (size_t)s * HC_D;
  const bf16_t* er = eproj + (size_t)wid * HC_D;
  float s0 = 0.f, s1 = 0.f;
#pragma unroll
  for (int j = 0; j < 4; ++j) {
    int ch = j * 32 + lane;
    s0 += qr[ch] * (kr[ch] + (float)er[ch]);
  }
#pragma unroll
  for (int j = 4; j < 8; ++j) {
    int ch = j * 32 + lane;
    s1 += qr[ch] * (kr[ch] + (float)er[ch]);
  }
#pragma unroll
  for (int o = 1; o < 32; o <<= 1) {
    s0 += __shfl_xor(s0, o);
    s1 += __shfl_xor(s1, o);
  }
  if (lane < 2) {
    float a = (lane == 0 ? s0 : s1) * 0.08838834764831845f;  // 1/sqrt(128)
    alpha[(size_t)wid * N_HEAD + lane] = a;
    atomicMax(&amax[d * N_HEAD + lane], enc_f32(a));
  }
}

// ---------------------------------------------------------------------------
// Pass 2: ea = exp(alpha - amax[dst]); denom[dst] += ea;
//         S[dst,:] += ea * (v[src,:] + e[e,:])   (normalize later per node)
// ---------------------------------------------------------------------------
__global__ void attn_scatter_kernel(const int* __restrict__ ei,
                                    const float* __restrict__ alpha,
                                    const unsigned* __restrict__ amax,
                                    const float* __restrict__ vmat,
                                    const bf16_t* __restrict__ eproj,
                                    float* __restrict__ denom,
                                    float* __restrict__ S) {
  int wid  = (int)((blockIdx.x * blockDim.x + threadIdx.x) >> 5);
  int lane = threadIdx.x & 31;
  if (wid >= N_EDGES) return;
  int s = ei[wid], d = ei[N_EDGES + wid];
  float ea0 = expf(alpha[(size_t)wid * N_HEAD + 0] -
                   dec_f32(amax[d * N_HEAD + 0]));
  float ea1 = expf(alpha[(size_t)wid * N_HEAD + 1] -
                   dec_f32(amax[d * N_HEAD + 1]));
  if (lane == 0) atomicAdd(&denom[d * N_HEAD + 0], ea0);
  if (lane == 1) atomicAdd(&denom[d * N_HEAD + 1], ea1);
  const float*  vr = vmat  + (size_t)s * HC_D;
  const bf16_t* er = eproj + (size_t)wid * HC_D;
  float* Sd = S + (size_t)d * HC_D;
#pragma unroll
  for (int j = 0; j < 8; ++j) {
    int ch = j * 32 + lane;
    float ea = (j < 4) ? ea0 : ea1;
    atomicAdd(&Sd[ch], ea * (vr[ch] + (float)er[ch]));
  }
}

// ---------------------------------------------------------------------------
// Node finalize: out = h + skip + S/(denom+1e-16), LayerNorm, optional SiLU.
// ---------------------------------------------------------------------------
__global__ void node_finalize_kernel(const float* __restrict__ hin,
                                     const float* __restrict__ sk,
                                     const float* __restrict__ S,
                                     const float* __restrict__ denom,
                                     const float* __restrict__ g,
                                     const float* __restrict__ b,
                                     float* __restrict__ hout,
                                     bf16_t* __restrict__ hout_b,
                                     int apply_silu) {
  int n = blockIdx.x;
  int ch = threadIdx.x;                 // 0..255
  int head = ch >> 7;
  size_t off = (size_t)n * HC_D + ch;
  float val = hin[off] + sk[off] + S[off] / (denom[n * N_HEAD + head] + 1e-16f);

  __shared__ float red[HC_D];
  red[ch] = val; __syncthreads();
  for (int o = 128; o > 0; o >>= 1) { if (ch < o) red[ch] += red[ch + o]; __syncthreads(); }
  float mu = red[0] * (1.f / HC_D); __syncthreads();
  float dv = val - mu;
  red[ch] = dv * dv; __syncthreads();
  for (int o = 128; o > 0; o >>= 1) { if (ch < o) red[ch] += red[ch + o]; __syncthreads(); }
  float var = red[0] * (1.f / HC_D);

  float y = dv * rsqrtf(var + 1e-5f) * g[ch] + b[ch];
  if (apply_silu) y = y / (1.f + expf(-y));
  hout[off] = y;
  if (hout_b) hout_b[off] = (bf16_t)y;
}

// ---------------------------------------------------------------------------
// Host orchestration
// ---------------------------------------------------------------------------
static inline void launch_gemm(const bf16_t* A, int lda, const bf16_t* Bp,
                               const float* bias, float* Cf, bf16_t* Cb,
                               int M, int K, hipStream_t stream) {
  int total  = (M / 32) * (HC_D / 64);
  int blocks = (total + 7) / 8;          // 8 waves per 256-thread block
  if (K == 128)
    gemm_wmma_bf16<128><<<blocks, 256, 0, stream>>>(A, lda, Bp, bias, Cf, Cb, M);
  else if (K == 224)
    gemm_wmma_bf16<224><<<blocks, 256, 0, stream>>>(A, lda, Bp, bias, Cf, Cb, M);
  else
    gemm_wmma_bf16<256><<<blocks, 256, 0, stream>>>(A, lda, Bp, bias, Cf, Cb, M);
}

extern "C" void kernel_launch(void* const* d_in, const int* in_sizes, int n_in,
                              void* d_out, int out_size, void* d_ws, size_t ws_size,
                              hipStream_t stream) {
  const float* x           = (const float*)d_in[0];
  const float* last_update = (const float*)d_in[1];
  const int*   edge_index  = (const int*)  d_in[2];
  const float* t           = (const float*)d_in[3];
  const float* msg         = (const float*)d_in[4];
  const float* time_w      = (const float*)d_in[5];
  const float* time_b      = (const float*)d_in[6];
  const float* lin_w       = (const float*)d_in[7];
  const float* lin_b       = (const float*)d_in[8];
  const float* Wq          = (const float*)d_in[9];
  const float* bq          = (const float*)d_in[10];
  const float* Wk          = (const float*)d_in[11];
  const float* bk          = (const float*)d_in[12];
  const float* Wv          = (const float*)d_in[13];
  const float* bv          = (const float*)d_in[14];
  const float* We          = (const float*)d_in[15];
  const float* Wsk         = (const float*)d_in[16];
  const float* bsk         = (const float*)d_in[17];
  const float* ln_g        = (const float*)d_in[18];
  const float* ln_b        = (const float*)d_in[19];
  float* out = (float*)d_out;
  (void)in_sizes; (void)n_in; (void)out_size; (void)ws_size;

  // ---- workspace carve (~450 MB total) ----
  char* p = (char*)d_ws;
  auto carve = [&](size_t bytes) -> char* {
    char* r = p; p += (bytes + 255) & ~(size_t)255; return r;
  };
  bf16_t*   eattr = (bf16_t*)  carve((size_t)N_EDGES * EDP_D * 2);  // 143 MB
  bf16_t*   eproj = (bf16_t*)  carve((size_t)N_EDGES * HC_D * 2);   // 164 MB
  float*    h     = (float*)   carve((size_t)N_NODES * HC_D * 4);
  bf16_t*   hb    = (bf16_t*)  carve((size_t)N_NODES * HC_D * 2);
  float*    qbuf  = (float*)   carve((size_t)N_NODES * HC_D * 4);
  float*    kbuf  = (float*)   carve((size_t)N_NODES * HC_D * 4);
  float*    vbuf  = (float*)   carve((size_t)N_NODES * HC_D * 4);
  float*    skbuf = (float*)   carve((size_t)N_NODES * HC_D * 4);
  float*    Sbuf  = (float*)   carve((size_t)N_NODES * HC_D * 4);
  float*    alpha = (float*)   carve((size_t)N_EDGES * N_HEAD * 4);
  unsigned* amax  = (unsigned*)carve((size_t)N_NODES * N_HEAD * 4);
  float*    denom = (float*)   carve((size_t)N_NODES * N_HEAD * 4);
  bf16_t*   xb    = (bf16_t*)  carve((size_t)N_NODES * MEM_D * 2);
  bf16_t*   linwp = (bf16_t*)  carve((size_t)MEM_D * HC_D * 2);
  bf16_t*   Wqp   = (bf16_t*)  carve((size_t)N_LAYER * IN_D * HC_D * 2);
  bf16_t*   Wkp   = (bf16_t*)  carve((size_t)N_LAYER * IN_D * HC_D * 2);
  bf16_t*   Wvp   = (bf16_t*)  carve((size_t)N_LAYER * IN_D * HC_D * 2);
  bf16_t*   Wskp  = (bf16_t*)  carve((size_t)N_LAYER * IN_D * HC_D * 2);
  bf16_t*   Wep   = (bf16_t*)  carve((size_t)N_LAYER * EDP_D * HC_D * 2);

  // ---- weight pre-swizzle into WMMA-B fragment order ----
  auto pack = [&](const float* src, bf16_t* dst, int Ksrc, int Kpad, int Nc) {
    int n = Kpad * Nc;
    pack_weight_kernel<<<(n + 255) / 256, 256, 0, stream>>>(src, dst, Ksrc, Kpad, Nc);
  };
  pack(lin_w, linwp, MEM_D, MEM_D, HC_D);
  // [3,256,256] packs as one K=768 pass (layer blocks land contiguously)
  pack(Wq,  Wqp,  N_LAYER * IN_D, N_LAYER * IN_D, HC_D);
  pack(Wk,  Wkp,  N_LAYER * IN_D, N_LAYER * IN_D, HC_D);
  pack(Wv,  Wvp,  N_LAYER * IN_D, N_LAYER * IN_D, HC_D);
  pack(Wsk, Wskp, N_LAYER * IN_D, N_LAYER * IN_D, HC_D);
  for (int l = 0; l < N_LAYER; ++l)
    pack(We + (size_t)l * ED_D * HC_D, Wep + (size_t)l * EDP_D * HC_D,
         ED_D, EDP_D, HC_D);

  {
    int n = N_NODES * MEM_D;
    f32_to_bf16_kernel<<<(n + 255) / 256, 256, 0, stream>>>(x, xb, n);
  }

  // ---- edge features (shared by all 3 layers) ----
  edge_feat_kernel<<<N_EDGES, EDP_D, 0, stream>>>(
      edge_index, last_update, t, msg, time_w, time_b, eattr);

  // ---- input projection: h0 = x @ lin_w + lin_b ----
  launch_gemm(xb, MEM_D, linwp, lin_b, h, hb, N_NODES, MEM_D, stream);

  // ---- 3 TransformerConv layers ----
  for (int l = 0; l < N_LAYER; ++l) {
    int nInit = N_NODES * HC_D;
    init_seg_kernel<<<(nInit + 255) / 256, 256, 0, stream>>>(Sbuf, denom, amax);

    size_t wofs = (size_t)l * IN_D * HC_D;
    launch_gemm(hb, HC_D, Wqp  + wofs, bq  + l * HC_D, qbuf,  nullptr,
                N_NODES, IN_D, stream);
    launch_gemm(hb, HC_D, Wkp  + wofs, bk  + l * HC_D, kbuf,  nullptr,
                N_NODES, IN_D, stream);
    launch_gemm(hb, HC_D, Wvp  + wofs, bv  + l * HC_D, vbuf,  nullptr,
                N_NODES, IN_D, stream);
    launch_gemm(hb, HC_D, Wskp + wofs, bsk + l * HC_D, skbuf, nullptr,
                N_NODES, IN_D, stream);

    // edge projection (dominant GEMM): [320k,224] @ [224,256] -> bf16
    launch_gemm(eattr, EDP_D, Wep + (size_t)l * EDP_D * HC_D, nullptr,
                nullptr, eproj, N_EDGES, EDP_D, stream);

    attn_alpha_kernel<<<N_EDGES / 8, 256, 0, stream>>>(
        edge_index, qbuf, kbuf, eproj, alpha, amax);
    attn_scatter_kernel<<<N_EDGES / 8, 256, 0, stream>>>(
        edge_index, alpha, amax, vbuf, eproj, denom, Sbuf);

    bool last = (l == N_LAYER - 1);
    node_finalize_kernel<<<N_NODES, HC_D, 0, stream>>>(
        h, skbuf, Sbuf, denom, ln_g + l * IN_D, ln_b + l * IN_D,
        last ? out : h, last ? nullptr : hb, last ? 0 : 1);
  }
}
